// EdgeUpdate_29661044146684
// MI455X (gfx1250) — matrix operand order repaired and checked by reading, not compiled, not run on hardware
//
#include <hip/hip_runtime.h>
#include <hip/hip_bf16.h>

// ---------------------------------------------------------------------------
// Problem constants (from reference setup_inputs)
// ---------------------------------------------------------------------------
#define BB   4
#define AT   48
#define NBR  20
#define FN   64
#define FE   64

#define R2   (BB*AT*NBR)        // 3840 rows, two-body
#define K2A  (2*FN)             // 128
#define N2A  (8*FN)             // 512
#define K2B  N2A                // 512
#define N2B  (2*FE)             // 128

#define R3   (BB*AT*NBR*NBR)    // 76800 rows, three-body
#define K3A  (3*FN+2*FE)        // 320
#define N3A  (6*FN+4*FE)        // 640
#define K3B  N3A                // 640
#define N3B  (2*FE)             // 128

#define WAVES_PER_BLOCK 8

typedef __attribute__((ext_vector_type(16))) _Float16 v16h;
typedef __attribute__((ext_vector_type(8)))  unsigned int v8u;
typedef __attribute__((ext_vector_type(8)))  float    v8f;

// ---------------------------------------------------------------------------
// Small device helpers
// ---------------------------------------------------------------------------
__device__ __forceinline__ unsigned umin32(unsigned a, unsigned b) { return a < b ? a : b; }
__device__ __forceinline__ unsigned umax32(unsigned a, unsigned b) { return a > b ? a : b; }

// Order-preserving float <-> uint encoding (so min/max can use integer atomics)
__device__ __forceinline__ unsigned encf(float f) {
    unsigned b = __float_as_uint(f);
    return (b & 0x80000000u) ? ~b : (b | 0x80000000u);
}
__device__ __forceinline__ float decf(unsigned u) {
    return (u & 0x80000000u) ? __uint_as_float(u & 0x7FFFFFFFu) : __uint_as_float(~u);
}

__device__ __forceinline__ unsigned hash32(unsigned x) {
    x ^= x >> 16; x *= 0x7feb352du;
    x ^= x >> 15; x *= 0x846ca68bu;
    x ^= x >> 16; return x;
}

// Approximate standard normal (sum of 4 uniforms); exact Threefry stream is
// not reproducible on device, structure matches the reference.
__device__ __forceinline__ float approx_normal(unsigned seed) {
    unsigned h0 = hash32(seed);
    unsigned h1 = hash32(seed ^ 0x68bc21ebu);
    unsigned h2 = hash32(seed ^ 0x02e5be93u);
    unsigned h3 = hash32(seed ^ 0x967a889bu);
    const float k = 1.0f / 16777216.0f;
    float s = (h0 >> 8) * k + (h1 >> 8) * k + (h2 >> 8) * k + (h3 >> 8) * k - 2.0f;
    return s * 1.7320508f;
}

__device__ __forceinline__ float sigmoidf(float x) { return 1.0f / (1.0f + expf(-x)); }

// Block-wide min/max reduction (encoded uints) + one global atomic per block
__device__ __forceinline__ void block_reduce_minmax(unsigned lmin, unsigned lmax,
                                                    unsigned* smin, unsigned* smax,
                                                    unsigned* gmin, unsigned* gmax) {
    int t = threadIdx.x;
    smin[t] = lmin; smax[t] = lmax;
    __syncthreads();
    for (int s = blockDim.x >> 1; s > 0; s >>= 1) {
        if (t < s) {
            smin[t] = umin32(smin[t], smin[t + s]);
            smax[t] = umax32(smax[t], smax[t + s]);
        }
        __syncthreads();
    }
    if (t == 0) { atomicMin(gmin, smin[0]); atomicMax(gmax, smax[0]); }
}

// ---------------------------------------------------------------------------
// Gathered input element values (never materialized in f32)
// ---------------------------------------------------------------------------
__device__ __forceinline__ float c2_value(const float* __restrict__ node,
                                          const float* __restrict__ mask,
                                          const int* __restrict__ idx,
                                          int b, int a, int j, int f) {
    if (f < FN) return node[(b * AT + a) * FN + f];
    int jn = idx[(b * AT + a) * NBR + j];
    return node[(b * AT + jn) * FN + (f - FN)] * mask[(b * AT + a) * NBR + j];
}

__device__ __forceinline__ float c3_value(const float* __restrict__ node,
                                          const float* __restrict__ edge,
                                          const float* __restrict__ mask,
                                          const int* __restrict__ idx,
                                          int b, int a, int j, int k, int f) {
    int jn = idx[(b * AT + a) * NBR + j];
    if (f < FN)        return node[(b * AT + a) * FN + f];
    if (f < 2 * FN)    return node[(b * AT + jn) * FN + (f - FN)] * mask[(b * AT + a) * NBR + j];
    if (f < 3 * FN) {
        int kn = idx[(b * AT + jn) * NBR + k];
        return node[(b * AT + kn) * FN + (f - 2 * FN)];
    }
    if (f < 3 * FN + FE) return edge[((b * AT + a) * NBR + j) * FE + (f - 3 * FN)];
    return edge[((b * AT + jn) * NBR + k) * FE + (f - 3 * FN - FE)];
}

// ---------------------------------------------------------------------------
// Kernels
// ---------------------------------------------------------------------------
__global__ void init_stats_kernel(unsigned* __restrict__ stats) {
    int t = threadIdx.x;
    if (t < 8) stats[t] = (t & 1) ? 0u : 0xFFFFFFFFu;  // [even]=min sentinel, [odd]=max sentinel
}

__global__ __launch_bounds__(256)
void minmax_c2_kernel(const float* __restrict__ node, const float* __restrict__ mask,
                      const int* __restrict__ idx, unsigned* __restrict__ stats) {
    __shared__ unsigned smin[256], smax[256];
    unsigned lmin = 0xFFFFFFFFu, lmax = 0u;
    const long long total = (long long)R2 * K2A;
    for (long long i = (long long)blockIdx.x * blockDim.x + threadIdx.x; i < total;
         i += (long long)gridDim.x * blockDim.x) {
        int f = (int)(i % K2A); long long t = i / K2A;
        int j = (int)(t % NBR); t /= NBR;
        int a = (int)(t % AT);  int b = (int)(t / AT);
        unsigned e = encf(c2_value(node, mask, idx, b, a, j, f));
        lmin = umin32(lmin, e); lmax = umax32(lmax, e);
    }
    block_reduce_minmax(lmin, lmax, smin, smax, &stats[0], &stats[1]);
}

__global__ __launch_bounds__(256)
void minmax_c3_kernel(const float* __restrict__ node, const float* __restrict__ edge,
                      const float* __restrict__ mask, const int* __restrict__ idx,
                      unsigned* __restrict__ stats) {
    __shared__ unsigned smin[256], smax[256];
    unsigned lmin = 0xFFFFFFFFu, lmax = 0u;
    const long long total = (long long)R3 * K3A;
    for (long long i = (long long)blockIdx.x * blockDim.x + threadIdx.x; i < total;
         i += (long long)gridDim.x * blockDim.x) {
        int f = (int)(i % K3A); long long t = i / K3A;
        int k = (int)(t % NBR); t /= NBR;
        int j = (int)(t % NBR); t /= NBR;
        int a = (int)(t % AT);  int b = (int)(t / AT);
        unsigned e = encf(c3_value(node, edge, mask, idx, b, a, j, k, f));
        lmin = umin32(lmin, e); lmax = umax32(lmax, e);
    }
    block_reduce_minmax(lmin, lmax, smin, smax, &stats[4], &stats[5]);
}

template <typename T>
__global__ __launch_bounds__(256)
void minmax_arr_kernel(const T* __restrict__ x, long long total,
                       unsigned* __restrict__ gmin, unsigned* __restrict__ gmax) {
    __shared__ unsigned smin[256], smax[256];
    unsigned lmin = 0xFFFFFFFFu, lmax = 0u;
    for (long long i = (long long)blockIdx.x * blockDim.x + threadIdx.x; i < total;
         i += (long long)gridDim.x * blockDim.x) {
        unsigned e = encf((float)x[i]);
        lmin = umin32(lmin, e); lmax = umax32(lmax, e);
    }
    block_reduce_minmax(lmin, lmax, smin, smax, gmin, gmax);
}

__global__ __launch_bounds__(256)
void build_q2_kernel(const float* __restrict__ node, const float* __restrict__ mask,
                     const int* __restrict__ idx, const unsigned* __restrict__ stats,
                     unsigned char* __restrict__ q) {
    const float mn = decf(stats[0]), mx = decf(stats[1]);
    const float inv = 255.0f / (mx - mn);
    const long long total = (long long)R2 * K2A;
    for (long long i = (long long)blockIdx.x * blockDim.x + threadIdx.x; i < total;
         i += (long long)gridDim.x * blockDim.x) {
        int f = (int)(i % K2A); long long t = i / K2A;
        int j = (int)(t % NBR); t /= NBR;
        int a = (int)(t % AT);  int b = (int)(t / AT);
        float x = c2_value(node, mask, idx, b, a, j, f);
        q[i] = (unsigned char)fminf((x - mn) * inv, 255.0f);  // trunc like torch uint8 cast
    }
}

__global__ __launch_bounds__(256)
void build_q3_kernel(const float* __restrict__ node, const float* __restrict__ edge,
                     const float* __restrict__ mask, const int* __restrict__ idx,
                     const unsigned* __restrict__ stats, unsigned char* __restrict__ q) {
    const float mn = decf(stats[4]), mx = decf(stats[5]);
    const float inv = 255.0f / (mx - mn);
    const long long total = (long long)R3 * K3A;
    for (long long i = (long long)blockIdx.x * blockDim.x + threadIdx.x; i < total;
         i += (long long)gridDim.x * blockDim.x) {
        int f = (int)(i % K3A); long long t = i / K3A;
        int k = (int)(t % NBR); t /= NBR;
        int j = (int)(t % NBR); t /= NBR;
        int a = (int)(t % AT);  int b = (int)(t / AT);
        float x = c3_value(node, edge, mask, idx, b, a, j, k, f);
        q[i] = (unsigned char)fminf((x - mn) * inv, 255.0f);
    }
}

template <typename T>
__global__ __launch_bounds__(256)
void quant_arr_kernel(const T* __restrict__ x, unsigned char* __restrict__ q,
                      long long total, const unsigned* __restrict__ stats, int stage) {
    const float mn = decf(stats[2 * stage]), mx = decf(stats[2 * stage + 1]);
    const float inv = 255.0f / (mx - mn);
    for (long long i = (long long)blockIdx.x * blockDim.x + threadIdx.x; i < total;
         i += (long long)gridDim.x * blockDim.x) {
        q[i] = (unsigned char)fminf(((float)x[i] - mn) * inv, 255.0f);
    }
}

// Expand W (N x K fp32) into 8 per-bit noisy f16 copies, PRE-SCALED by 2^bit so
// the GEMM's A operand is a pure 0/1 matrix:  Wn_i = (W + n()*|W|*0.1) * 2^i
__global__ __launch_bounds__(256)
void gen_noisy_kernel(const float* __restrict__ W, _Float16* __restrict__ wn,
                      int N, int K, unsigned stageSeed) {
    const long long nk = (long long)N * K;
    const long long total = 8 * nk;
    for (long long i = (long long)blockIdx.x * blockDim.x + threadIdx.x; i < total;
         i += (long long)gridDim.x * blockDim.x) {
        int bit = (int)(i / nk);
        long long rem = i - (long long)bit * nk;
        float w = W[rem];
        unsigned seed = hash32(stageSeed * 0x9E3779B9u + (unsigned)bit * 0x85EBCA6Bu) ^ (unsigned)rem;
        float nz = approx_normal(seed) * fabsf(w) * 0.1f;
        wn[i] = (_Float16)((w + nz) * (float)(1 << bit));
    }
}

// rowsum of the bit-7 noisy weight (requant bias: mn * sum_k W7[n,k]).
// Stored W7 is pre-scaled by 2^7, so divide back out.
__global__ void rowsum_kernel(const _Float16* __restrict__ wn, float* __restrict__ rs,
                              int N, int K) {
    int n = blockIdx.x * blockDim.x + threadIdx.x;
    if (n >= N) return;
    const _Float16* p = wn + ((size_t)7 * N + n) * K;
    float s = 0.0f;
    for (int k = 0; k < K; ++k) s += (float)p[k];
    rs[n] = s * (1.0f / 128.0f);
}

// ---------------------------------------------------------------------------
// Bit-serial quantized GEMM on the WMMA path.
//   out[r,n] = requant( sum_bit  Bit_i(q)[r,:] @ (2^i * Wn_i)[n,:]^T )
// One wave -> a 32x64 output macro-tile (2 row-tiles x 4 n-tiles, 8 f32
// accumulators). Each B fragment feeds 2 WMMAs (halves B-operand bandwidth to
// 512 B/WMMA), each 0/1 A fragment feeds 4 WMMAs, and the 8 back-to-back
// V_WMMA_F32_16X16X32_F16 ops on distinct accumulators hide RAW latency.
// A fragment build: packed f16 pair = __umul24(((bytes>>bit)&0x10001), 0x3C00)
//   -> 3 integer VALU ops per 2 elements, no compares/cndmask.
// B operand per lane: 16 contiguous f16 from the noisy-weight row (K-major).
// ---------------------------------------------------------------------------
template <bool RELU, typename OutT>
__global__ __launch_bounds__(256)
void qgemm_wmma_kernel(const unsigned char* __restrict__ q,
                       const _Float16* __restrict__ wn,       // 8 x N x K, f16 (2^bit-scaled)
                       const float* __restrict__ rowsum,      // N
                       const unsigned* __restrict__ stats, int stage,
                       OutT* __restrict__ out,
                       int R, int K, int N) {
    const int lane = threadIdx.x & 31;
    const int wave = threadIdx.x >> 5;
    const int nGroups = N >> 6;  // groups of 4 n-tiles (64 output columns)
    const long long ntiles = (long long)(R >> 5) * nGroups;  // 32-row macro-tiles
    const long long tile = (long long)blockIdx.x * WAVES_PER_BLOCK + wave;
    if (tile >= ntiles) return;  // wave-uniform: EXEC stays all-ones for WMMA

    const int rt = (int)(tile / nGroups);
    const int ng = (int)(tile % nGroups);
    const int m = lane & 15;
    const int half = lane >> 4;

    const unsigned char* qrow0 = q + (size_t)(rt * 32 + m) * K;
    const unsigned char* qrow1 = q + (size_t)(rt * 32 + 16 + m) * K;
    const size_t bitStride = (size_t)N * K;  // f16 elements between bit planes
    const _Float16* wbase = wn + ((size_t)(ng * 64 + m)) * K + half * 16;

    v8f acc0 = {}, acc1 = {}, acc2 = {}, acc3 = {};
    v8f acc4 = {}, acc5 = {}, acc6 = {}, acc7 = {};
    const int ksteps = K >> 5;
    for (int ks = 0; ks < ksteps; ++ks) {
        const int kbase = ks << 5;
        unsigned long long lo0 = *(const unsigned long long*)(qrow0 + kbase + half * 16);
        unsigned long long hi0 = *(const unsigned long long*)(qrow0 + kbase + half * 16 + 8);
        unsigned long long lo1 = *(const unsigned long long*)(qrow1 + kbase + half * 16);
        unsigned long long hi1 = *(const unsigned long long*)(qrow1 + kbase + half * 16 + 8);
        if (ks + 1 < ksteps) {
            __builtin_prefetch(qrow0 + kbase + 32, 0, 3);
            __builtin_prefetch(qrow1 + kbase + 32, 0, 3);
        }

        // spread byte pairs into 16-bit lanes: x[p] = byte(2p) | byte(2p+1)<<16
        unsigned x0[8], x1[8];
#pragma unroll
        for (int p = 0; p < 4; ++p) {
            unsigned wl0 = (unsigned)(lo0 >> (16 * p));
            unsigned wh0 = (unsigned)(hi0 >> (16 * p));
            unsigned wl1 = (unsigned)(lo1 >> (16 * p));
            unsigned wh1 = (unsigned)(hi1 >> (16 * p));
            x0[p]     = (wl0 & 0xFFu) | ((wl0 & 0xFF00u) << 8);
            x0[p + 4] = (wh0 & 0xFFu) | ((wh0 & 0xFF00u) << 8);
            x1[p]     = (wl1 & 0xFFu) | ((wl1 & 0xFF00u) << 8);
            x1[p + 4] = (wh1 & 0xFFu) | ((wh1 & 0xFF00u) << 8);
        }

        const _Float16* wk = wbase + kbase;
#pragma unroll
        for (int bit = 0; bit < 8; ++bit) {
            v8u au0, au1;
#pragma unroll
            for (int p = 0; p < 8; ++p) {
                unsigned t0 = (x0[p] >> bit) & 0x00010001u;  // 0/1 per 16-bit lane
                unsigned t1 = (x1[p] >> bit) & 0x00010001u;
                au0[p] = __umul24(t0, 0x3C00u);              // packed f16 pair {0,1}
                au1[p] = __umul24(t1, 0x3C00u);
            }
            v16h av0 = __builtin_bit_cast(v16h, au0);
            v16h av1 = __builtin_bit_cast(v16h, au1);
            const _Float16* wb = wk + (size_t)bit * bitStride;
            v16h b0 = *(const v16h*)(wb);
            v16h b1 = *(const v16h*)(wb + (size_t)16 * K);
            v16h b2 = *(const v16h*)(wb + (size_t)32 * K);
            v16h b3 = *(const v16h*)(wb + (size_t)48 * K);
            acc0 = __builtin_amdgcn_wmma_f32_16x16x32_f16(false, av0, false, b0, (short)0, acc0, false, false);
            acc4 = __builtin_amdgcn_wmma_f32_16x16x32_f16(false, av1, false, b0, (short)0, acc4, false, false);
            acc1 = __builtin_amdgcn_wmma_f32_16x16x32_f16(false, av0, false, b1, (short)0, acc1, false, false);
            acc5 = __builtin_amdgcn_wmma_f32_16x16x32_f16(false, av1, false, b1, (short)0, acc5, false, false);
            acc2 = __builtin_amdgcn_wmma_f32_16x16x32_f16(false, av0, false, b2, (short)0, acc2, false, false);
            acc6 = __builtin_amdgcn_wmma_f32_16x16x32_f16(false, av1, false, b2, (short)0, acc6, false, false);
            acc3 = __builtin_amdgcn_wmma_f32_16x16x32_f16(false, av0, false, b3, (short)0, acc3, false, false);
            acc7 = __builtin_amdgcn_wmma_f32_16x16x32_f16(false, av1, false, b3, (short)0, acc7, false, false);
        }
    }

    // requant: z/255*(mx-mn) + mn*rowsum[n]  (+ optional ReLU)
    const float mn = decf(stats[2 * stage]);
    const float mx = decf(stats[2 * stage + 1]);
    const float sc = (mx - mn) * (1.0f / 255.0f);
    v8f accs[8] = {acc0, acc1, acc2, acc3, acc4, acc5, acc6, acc7};
#pragma unroll
    for (int rh = 0; rh < 2; ++rh) {
#pragma unroll
        for (int t4 = 0; t4 < 4; ++t4) {
            const int ncol = ng * 64 + t4 * 16 + m;
            const float bias = mn * rowsum[ncol];
#pragma unroll
            for (int v = 0; v < 8; ++v) {
                float c = accs[rh * 4 + t4][v] * sc + bias;
                if (RELU) c = fmaxf(c, 0.0f);
                const int r = rt * 32 + rh * 16 + half * 8 + v;
                out[(size_t)r * N + ncol] = (OutT)c;
            }
        }
    }
}

// two-body gate: two[r,f] = sigmoid(c[r,f]) * tanh(c[r,64+f])
__global__ void gate2_kernel(const float* __restrict__ c, float* __restrict__ two) {
    int i = blockIdx.x * blockDim.x + threadIdx.x;
    if (i >= R2 * FE) return;
    int r = i >> 6, f = i & 63;
    float g = c[(size_t)r * N2B + f];
    float e = c[(size_t)r * N2B + FE + f];
    two[i] = sigmoidf(g) * tanhf(e);
}

// three-body gate + masked sum over inner neighbor axis
__global__ void gate3_kernel(const float* __restrict__ c, const float* __restrict__ mask,
                             const int* __restrict__ idx, float* __restrict__ three) {
    int i = blockIdx.x * blockDim.x + threadIdx.x;
    if (i >= R2 * FE) return;
    int r = i >> 6, f = i & 63;
    int b = r / (AT * NBR), rem = r % (AT * NBR);
    int a = rem / NBR, j = rem % NBR;
    int jn = idx[(b * AT + a) * NBR + j];
    float s = 0.0f;
    for (int k = 0; k < NBR; ++k) {
        const float* row = c + (size_t)(r * NBR + k) * N3B;
        float mk = mask[(b * AT + jn) * NBR + k];
        s += sigmoidf(row[f]) * tanhf(row[FE + f]) * mk;
    }
    three[i] = s;
}

// per-feature batchnorm statistics over 3840 rows (biased var, like torch BN training)
__global__ __launch_bounds__(256)
void bn_stats_kernel(const float* __restrict__ x, float* __restrict__ meanOut,
                     float* __restrict__ istdOut) {
    __shared__ float sh[256], sh2[256];
    int f = blockIdx.x;  // 64 features
    float s = 0.0f, s2 = 0.0f;
    for (int r = threadIdx.x; r < R2; r += 256) {
        float v = x[(size_t)r * FE + f];
        s += v; s2 += v * v;
    }
    sh[threadIdx.x] = s; sh2[threadIdx.x] = s2;
    __syncthreads();
    for (int st = 128; st > 0; st >>= 1) {
        if (threadIdx.x < st) { sh[threadIdx.x] += sh[threadIdx.x + st]; sh2[threadIdx.x] += sh2[threadIdx.x + st]; }
        __syncthreads();
    }
    if (threadIdx.x == 0) {
        float m = sh[0] / (float)R2;
        float var = sh2[0] / (float)R2 - m * m;
        meanOut[f] = m;
        istdOut[f] = rsqrtf(var + 1e-5f);
    }
}

// out = relu(edge + bn(two) + bn(three))
__global__ void final_kernel(const float* __restrict__ edge, const float* __restrict__ two,
                             const float* __restrict__ three, const float* __restrict__ bn,
                             const float* __restrict__ g2, const float* __restrict__ b2,
                             const float* __restrict__ g3, const float* __restrict__ b3,
                             float* __restrict__ out) {
    int i = blockIdx.x * blockDim.x + threadIdx.x;
    if (i >= R2 * FE) return;
    int f = i & 63;
    float t2 = (two[i] - bn[f]) * bn[64 + f] * g2[f] + b2[f];
    float t3 = (three[i] - bn[128 + f]) * bn[192 + f] * g3[f] + b3[f];
    out[i] = fmaxf(edge[i] + t2 + t3, 0.0f);
}

// ---------------------------------------------------------------------------
// Host: stage the pipeline on `stream`
// ---------------------------------------------------------------------------
extern "C" void kernel_launch(void* const* d_in, const int* in_sizes, int n_in,
                              void* d_out, int out_size, void* d_ws, size_t ws_size,
                              hipStream_t stream) {
    const float* node = (const float*)d_in[0];
    const float* edge = (const float*)d_in[1];
    const float* mask = (const float*)d_in[2];
    const float* c_two    = (const float*)d_in[3];
    const float* c_two2   = (const float*)d_in[4];
    const float* c_three  = (const float*)d_in[5];
    const float* c_three2 = (const float*)d_in[6];
    const float* bn2_g = (const float*)d_in[7];
    const float* bn2_b = (const float*)d_in[8];
    const float* bn3_g = (const float*)d_in[9];
    const float* bn3_b = (const float*)d_in[10];
    const int*   nbr_idx = (const int*)d_in[11];
    float* out = (float*)d_out;

    // workspace layout (256B aligned blocks)
    size_t cur = 0;
    auto alloc = [&](size_t bytes) { size_t o = cur; cur = (cur + bytes + 255) & ~(size_t)255; return o; };
    char* ws = (char*)d_ws;

    size_t o_stats = alloc(64 * 4);                 // 8 used u32 slots (min/max per stage 0..3)
    size_t o_bn    = alloc(256 * 4);                // mean2, istd2, mean3, istd3
    size_t o_rs2a  = alloc((size_t)N2A * 4);
    size_t o_rs2b  = alloc((size_t)N2B * 4);
    size_t o_rs3a  = alloc((size_t)N3A * 4);
    size_t o_rs3b  = alloc((size_t)N3B * 4);
    size_t o_w2a   = alloc((size_t)8 * N2A * K2A * 2);
    size_t o_w2b   = alloc((size_t)8 * N2B * K2B * 2);
    size_t o_w3a   = alloc((size_t)8 * N3A * K3A * 2);
    size_t o_w3b   = alloc((size_t)8 * N3B * K3B * 2);
    size_t o_q2    = alloc((size_t)R2 * K2A);
    size_t o_a2    = alloc((size_t)R2 * N2A * 4);   // f32
    size_t o_q2b   = alloc((size_t)R2 * K2B);
    size_t o_c2o   = alloc((size_t)R2 * N2B * 4);
    size_t o_q3    = alloc((size_t)R3 * K3A);
    size_t o_a3    = alloc((size_t)R3 * N3A * 2);   // f16 intermediate (halves ws traffic)
    size_t o_q3b   = alloc((size_t)R3 * K3B);
    size_t o_c3o   = alloc((size_t)R3 * N3B * 4);
    size_t o_two   = alloc((size_t)R2 * FE * 4);
    size_t o_three = alloc((size_t)R2 * FE * 4);
    (void)ws_size; (void)in_sizes; (void)n_in; (void)out_size;

    unsigned*      stats = (unsigned*)(ws + o_stats);
    float*         bn    = (float*)(ws + o_bn);
    float*         rs2a  = (float*)(ws + o_rs2a);
    float*         rs2b  = (float*)(ws + o_rs2b);
    float*         rs3a  = (float*)(ws + o_rs3a);
    float*         rs3b  = (float*)(ws + o_rs3b);
    _Float16*      w2a   = (_Float16*)(ws + o_w2a);
    _Float16*      w2b   = (_Float16*)(ws + o_w2b);
    _Float16*      w3a   = (_Float16*)(ws + o_w3a);
    _Float16*      w3b   = (_Float16*)(ws + o_w3b);
    unsigned char* q2    = (unsigned char*)(ws + o_q2);
    float*         a2    = (float*)(ws + o_a2);
    unsigned char* q2b   = (unsigned char*)(ws + o_q2b);
    float*         c2o   = (float*)(ws + o_c2o);
    unsigned char* q3    = (unsigned char*)(ws + o_q3);
    _Float16*      a3    = (_Float16*)(ws + o_a3);
    unsigned char* q3b   = (unsigned char*)(ws + o_q3b);
    float*         c3o   = (float*)(ws + o_c3o);
    float*         two   = (float*)(ws + o_two);
    float*         three = (float*)(ws + o_three);

    hipLaunchKernelGGL(init_stats_kernel, dim3(1), dim3(32), 0, stream, stats);

    // ---------------- two-body path ----------------
    hipLaunchKernelGGL(minmax_c2_kernel, dim3(480), dim3(256), 0, stream, node, mask, nbr_idx, stats);
    hipLaunchKernelGGL(gen_noisy_kernel, dim3(1024), dim3(256), 0, stream, c_two, w2a, N2A, K2A, 0u);
    hipLaunchKernelGGL(rowsum_kernel, dim3((N2A + 255) / 256), dim3(256), 0, stream, w2a, rs2a, N2A, K2A);
    hipLaunchKernelGGL(build_q2_kernel, dim3(1920), dim3(256), 0, stream, node, mask, nbr_idx, stats, q2);
    hipLaunchKernelGGL((qgemm_wmma_kernel<true, float>),
                       dim3((R2 / 32) * (N2A / 64) / WAVES_PER_BLOCK), dim3(256), 0, stream,
                       q2, w2a, rs2a, stats, 0, a2, R2, K2A, N2A);

    hipLaunchKernelGGL((minmax_arr_kernel<float>), dim3(2048), dim3(256), 0, stream,
                       a2, (long long)R2 * N2A, &stats[2], &stats[3]);
    hipLaunchKernelGGL(gen_noisy_kernel, dim3(1024), dim3(256), 0, stream, c_two2, w2b, N2B, K2B, 1u);
    hipLaunchKernelGGL(rowsum_kernel, dim3((N2B + 255) / 256), dim3(256), 0, stream, w2b, rs2b, N2B, K2B);
    hipLaunchKernelGGL((quant_arr_kernel<float>), dim3(2048), dim3(256), 0, stream,
                       a2, q2b, (long long)R2 * K2B, stats, 1);
    hipLaunchKernelGGL((qgemm_wmma_kernel<false, float>),
                       dim3((R2 / 32) * (N2B / 64) / WAVES_PER_BLOCK), dim3(256), 0, stream,
                       q2b, w2b, rs2b, stats, 1, c2o, R2, K2B, N2B);
    hipLaunchKernelGGL(gate2_kernel, dim3((R2 * FE + 255) / 256), dim3(256), 0, stream, c2o, two);

    // ---------------- three-body path ----------------
    hipLaunchKernelGGL(minmax_c3_kernel, dim3(4096), dim3(256), 0, stream, node, edge, mask, nbr_idx, stats);
    hipLaunchKernelGGL(gen_noisy_kernel, dim3(2048), dim3(256), 0, stream, c_three, w3a, N3A, K3A, 2u);
    hipLaunchKernelGGL(rowsum_kernel, dim3((N3A + 255) / 256), dim3(256), 0, stream, w3a, rs3a, N3A, K3A);
    hipLaunchKernelGGL(build_q3_kernel, dim3(8192), dim3(256), 0, stream, node, edge, mask, nbr_idx, stats, q3);
    hipLaunchKernelGGL((qgemm_wmma_kernel<true, _Float16>),
                       dim3((R3 / 32) * (N3A / 64) / WAVES_PER_BLOCK), dim3(256), 0, stream,
                       q3, w3a, rs3a, stats, 2, a3, R3, K3A, N3A);

    hipLaunchKernelGGL((minmax_arr_kernel<_Float16>), dim3(8192), dim3(256), 0, stream,
                       a3, (long long)R3 * N3A, &stats[6], &stats[7]);
    hipLaunchKernelGGL(gen_noisy_kernel, dim3(2048), dim3(256), 0, stream, c_three2, w3b, N3B, K3B, 3u);
    hipLaunchKernelGGL(rowsum_kernel, dim3((N3B + 255) / 256), dim3(256), 0, stream, w3b, rs3b, N3B, K3B);
    hipLaunchKernelGGL((quant_arr_kernel<_Float16>), dim3(8192), dim3(256), 0, stream,
                       a3, q3b, (long long)R3 * K3B, stats, 3);
    hipLaunchKernelGGL((qgemm_wmma_kernel<false, float>),
                       dim3((R3 / 32) * (N3B / 64) / WAVES_PER_BLOCK), dim3(256), 0, stream,
                       q3b, w3b, rs3b, stats, 3, c3o, R3, K3B, N3B);
    hipLaunchKernelGGL(gate3_kernel, dim3((R2 * FE + 255) / 256), dim3(256), 0, stream,
                       c3o, mask, nbr_idx, three);

    // ---------------- batchnorm + final ----------------
    hipLaunchKernelGGL(bn_stats_kernel, dim3(FE), dim3(256), 0, stream, two, &bn[0], &bn[64]);
    hipLaunchKernelGGL(bn_stats_kernel, dim3(FE), dim3(256), 0, stream, three, &bn[128], &bn[192]);
    hipLaunchKernelGGL(final_kernel, dim3((R2 * FE + 255) / 256), dim3(256), 0, stream,
                       edge, two, three, bn, bn2_g, bn2_b, bn3_g, bn3_b, out);
}
